// CausalSelfAttention_12678743458293
// MI455X (gfx1250) — matrix operand order repaired and checked by reading, not compiled, not run on hardware
//
#include <hip/hip_runtime.h>
#include <hip/hip_bf16.h>

// ---------------------------------------------------------------------------
// Causal self-attention for MI455X (gfx1250), wave32, bf16 WMMA everywhere.
// Pipeline: cvt(x)->bf16, cvt+transpose(W*)->bf16 ; Q/K/V = GEMM ; flash attn ;
// out = GEMM(Y, Wp) + bp  (f32 out).
// GEMM uses double-buffered GLOBAL_LOAD_ASYNC_TO_LDS_B128 staging (ASYNCcnt).
// ---------------------------------------------------------------------------

typedef __attribute__((ext_vector_type(16))) __bf16 v16bf;
typedef __attribute__((ext_vector_type(8)))  float  v8f;

#define LDT 72          // LDS row stride in bf16 elements (64 + 8 pad, 144B: 16B aligned)

#if __has_builtin(__builtin_amdgcn_s_wait_asynccnt)
#define WAIT_ASYNC(N) __builtin_amdgcn_s_wait_asynccnt(N)
#else
#define WAIT_ASYNC(N) asm volatile("s_wait_asynccnt " #N ::: "memory")
#endif

// Low 32 bits of a generic (flat) pointer to a __shared__ object are the LDS
// byte offset (ISA 10.2: LDS aperture).
__device__ __forceinline__ unsigned int lds_off(const void* p) {
    return (unsigned int)(size_t)p;
}

// Per-lane async copy of 16 bytes global -> LDS (GV mode, ASYNCcnt-tracked).
__device__ __forceinline__ void async_ld_b128(unsigned int lds, const void* g) {
    asm volatile("global_load_async_to_lds_b128 %0, %1, off"
                 :: "v"(lds), "v"((unsigned long long)(size_t)g)
                 : "memory");
}

__device__ __forceinline__ unsigned short f32_to_bf16u(float f) {
    unsigned int u = __float_as_uint(f);
    u = (u + 0x7FFFu + ((u >> 16) & 1u)) >> 16;   // round-to-nearest-even
    return (unsigned short)u;
}

__device__ __forceinline__ v8f wmma_bf16(v16bf a, v16bf b, v8f c) {
    // D(16x16,f32) = A(16x32,bf16) x B(32x16,bf16) + C
    return __builtin_amdgcn_wmma_f32_16x16x32_bf16(false, a, false, b, (short)0, c,
                                                   false, false);
}

// A fragment: 16x32 tile, row-major in LDS with stride ld.
// Layout: lane row = lane&15 ; lanes 0-15 hold K=0..15, lanes 16-31 hold K=16..31.
__device__ __forceinline__ v16bf load_a_frag(const __bf16* __restrict__ t, int ld) {
    const int lane = threadIdx.x & 31;
    const __bf16* p = t + (lane & 15) * ld + ((lane >> 4) << 4);
    v16bf a;
#pragma unroll
    for (int e = 0; e < 16; ++e) a[e] = p[e];
    return a;
}

// B fragment: 32x16 (KxN) tile stored N-major in LDS (row = N, col = K, stride ld).
// Layout (contiguous k, same as A): lane N = lane&15 ; lanes 0-15 hold K=0..15, lanes 16-31 hold K=16..31.
__device__ __forceinline__ v16bf load_b_frag(const __bf16* __restrict__ t, int ld) {
    const int lane = threadIdx.x & 31;
    const __bf16* p = t + (lane & 15) * ld + ((lane >> 4) << 4);
    v16bf b;
#pragma unroll
    for (int e = 0; e < 16; ++e) b[e] = p[e];
    return b;
}

__device__ __forceinline__ void store_out(float* p, float v)          { *p = v; }
__device__ __forceinline__ void store_out(unsigned short* p, float v) { *p = f32_to_bf16u(v); }

// --------------------------- conversion kernels ----------------------------

__global__ __launch_bounds__(256) void k_cvt_bf16(const float* __restrict__ in,
                                                  unsigned short* __restrict__ out, int n) {
    int i = blockIdx.x * 256 + threadIdx.x;
    if (i < n) out[i] = f32_to_bf16u(in[i]);
}

// W (KxN, f32, row-major)  ->  Wt (NxK, bf16)
__global__ __launch_bounds__(256) void k_cvt_transpose(const float* __restrict__ W,
                                                       unsigned short* __restrict__ Wt,
                                                       int K, int N) {
    int i = blockIdx.x * 256 + threadIdx.x;
    if (i < K * N) {
        int k = i / N, n = i - k * N;
        Wt[(size_t)n * K + k] = f32_to_bf16u(W[i]);
    }
}

// ------------------------------- GEMM --------------------------------------
// C(MxN) = A(MxK,bf16) @ B(KxN) + bias, with B given transposed (Bt: NxK, bf16).
// BM=128 BN=64 BK=64, 256 threads (8 waves), wave tile 32x32 (2x2 WMMA frags).
// Double-buffered async staging: 6 b128 asyncs/thread/chunk; while computing
// chunk i, chunk i+1's asyncs are outstanding -> s_wait_asynccnt 6 (in-order
// completion guarantees chunk i has landed).

template <typename OutT>
__global__ __launch_bounds__(256)
void k_gemm(const unsigned short* __restrict__ A, const unsigned short* __restrict__ Bt,
            const float* __restrict__ bias, OutT* __restrict__ Cout,
            int Mdim, int Ndim, int Kdim) {
    __shared__ __align__(16) unsigned short As[2][128 * LDT];   // 2 x 18 KB
    __shared__ __align__(16) unsigned short Bs[2][64 * LDT];    // 2 x  9 KB

    const int tid  = threadIdx.x;
    const int lane = tid & 31;
    const int w    = tid >> 5;
    const int wm   = w & 3;          // 4 wave rows  (32 rows each)
    const int wn   = w >> 2;         // 2 wave cols  (32 cols each)
    const int m0   = blockIdx.y * 128;
    const int n0   = blockIdx.x * 64;

    const int arow = tid >> 1, acol = (tid & 1) * 32;   // A stage: 128x64
    const int brow = tid >> 2, bcol = (tid & 3) * 16;   // B stage: 64x64

    auto stage = [&](int buf, int k0) {
        const unsigned short* gA = A + (size_t)(m0 + arow) * Kdim + k0 + acol;
        unsigned int la = lds_off(&As[buf][arow * LDT + acol]);
        async_ld_b128(la,      gA);
        async_ld_b128(la + 16, gA + 8);
        async_ld_b128(la + 32, gA + 16);
        async_ld_b128(la + 48, gA + 24);
        const unsigned short* gB = Bt + (size_t)(n0 + brow) * Kdim + k0 + bcol;
        unsigned int lb = lds_off(&Bs[buf][brow * LDT + bcol]);
        async_ld_b128(lb,      gB);
        async_ld_b128(lb + 16, gB + 8);
    };

    v8f acc[2][2] = {};
    const int nk = Kdim >> 6;

    stage(0, 0);
    for (int i = 0; i < nk; ++i) {
        const int cur = i & 1;
        const bool more = (i + 1 < nk);
        if (more) stage(1 - cur, (i + 1) * 64);          // prefetch next chunk to LDS
        if (i + 2 < nk) {                                 // and chunk i+2 to L2
            __builtin_prefetch((const void*)(A + (size_t)(m0 + arow) * Kdim + (i + 2) * 64 + acol), 0, 1);
            __builtin_prefetch((const void*)(Bt + (size_t)(n0 + brow) * Kdim + (i + 2) * 64 + bcol), 0, 1);
        }
        if (more) { WAIT_ASYNC(6); } else { WAIT_ASYNC(0); }   // chunk i landed
        __syncthreads();

        const __bf16* as = (const __bf16*)As[cur];
        const __bf16* bs = (const __bf16*)Bs[cur];
#pragma unroll
        for (int kk = 0; kk < 64; kk += 32) {
            v16bf a0 = load_a_frag(as + (wm * 32 +  0) * LDT + kk, LDT);
            v16bf a1 = load_a_frag(as + (wm * 32 + 16) * LDT + kk, LDT);
            v16bf b0 = load_b_frag(bs + (wn * 32 +  0) * LDT + kk, LDT);
            v16bf b1 = load_b_frag(bs + (wn * 32 + 16) * LDT + kk, LDT);
            acc[0][0] = wmma_bf16(a0, b0, acc[0][0]);
            acc[0][1] = wmma_bf16(a0, b1, acc[0][1]);
            acc[1][0] = wmma_bf16(a1, b0, acc[1][0]);
            acc[1][1] = wmma_bf16(a1, b1, acc[1][1]);
        }
        __syncthreads();   // all waves done with buf[cur] before it is restaged
    }

    const int hl = lane >> 4;      // C frag: row = r + 8*hl, col = lane&15
    const int nl = lane & 15;
#pragma unroll
    for (int i = 0; i < 2; ++i)
#pragma unroll
        for (int j = 0; j < 2; ++j) {
            const int ncol = n0 + wn * 32 + j * 16 + nl;
            const float bv = bias[ncol];
#pragma unroll
            for (int r = 0; r < 8; ++r) {
                const int mrow = m0 + wm * 32 + i * 16 + r + 8 * hl;
                store_out(&Cout[(size_t)mrow * Ndim + ncol], acc[i][j][r] + bv);
            }
        }
}

// --------------------------- flash attention -------------------------------
// One block = (b, h, 64-query tile). 128 threads = 4 waves; wave owns 16 q-rows.
// Q/K/V are bf16 in (B,T,C) layout, C = H*64.

__global__ __launch_bounds__(128)
void k_attn(const unsigned short* __restrict__ Qg, const unsigned short* __restrict__ Kg,
            const unsigned short* __restrict__ Vg, unsigned short* __restrict__ Yg) {
    __shared__ __align__(16) unsigned short Qs [64 * LDT];
    __shared__ __align__(16) unsigned short Ks [64 * LDT];
    __shared__ __align__(16) unsigned short Vts[64 * LDT];   // transposed: [d][key]
    __shared__ __align__(16) unsigned short Ps [64 * LDT];   // wave-private 16-row slabs

    const int tid  = threadIdx.x;
    const int lane = tid & 31;
    const int w    = tid >> 5;
    const int hl   = lane >> 4;
    const int nl   = lane & 15;

    const int qt = blockIdx.x;
    const int b  = blockIdx.y >> 4;
    const int h  = blockIdx.y & 15;
    const int q0 = qt * 64;
    const float scale = 0.125f;        // 1/sqrt(64)

    const int srow = tid >> 1, scol = (tid & 1) * 32;   // 64x64 stage mapping

    {   // stage Q tile (64 rows x 64 d) via async-to-LDS
        const unsigned short* g =
            Qg + ((size_t)(b * 2048 + q0 + srow)) * 1024 + h * 64 + scol;
        unsigned int l = lds_off(&Qs[srow * LDT + scol]);
        async_ld_b128(l,      g);
        async_ld_b128(l + 16, g + 8);
        async_ld_b128(l + 32, g + 16);
        async_ld_b128(l + 48, g + 24);
    }

    float row_m[8], row_l[8];
#pragma unroll
    for (int r = 0; r < 8; ++r) { row_m[r] = -1e30f; row_l[r] = 0.0f; }
    v8f o[4] = {};

    for (int kt = 0; kt <= qt; ++kt) {
        const int k0 = kt * 64;
        __syncthreads();                       // all waves done reading Ks/Vts
        {   // stage K tile, key-major (== N-major for S=Q·K^T) via async-to-LDS
            const unsigned short* g =
                Kg + ((size_t)(b * 2048 + k0 + srow)) * 1024 + h * 64 + scol;
            unsigned int l = lds_off(&Ks[srow * LDT + scol]);
            async_ld_b128(l,      g);
            async_ld_b128(l + 16, g + 8);
            async_ld_b128(l + 32, g + 16);
            async_ld_b128(l + 48, g + 24);
        }
        {   // stage V transposed: Vts[d][key]  (N-major for O=P·V)
            const uint4* g = (const uint4*)(Vg + ((size_t)(b * 2048 + k0 + srow)) * 1024 + h * 64 + scol);
            unsigned short tmp[32];
            *(uint4*)(tmp +  0) = g[0]; *(uint4*)(tmp +  8) = g[1];
            *(uint4*)(tmp + 16) = g[2]; *(uint4*)(tmp + 24) = g[3];
#pragma unroll
            for (int e = 0; e < 32; ++e) Vts[(scol + e) * LDT + srow] = tmp[e];
        }
        WAIT_ASYNC(0);                         // K (and Q on iter 0) landed
        __syncthreads();

        // S = Q(16x64) @ K^T  -> 4 column fragments
        v8f s4[4] = {};
#pragma unroll
        for (int kk = 0; kk < 64; kk += 32) {
            v16bf aq = load_a_frag((const __bf16*)Qs + (w * 16) * LDT + kk, LDT);
#pragma unroll
            for (int j = 0; j < 4; ++j) {
                v16bf bk = load_b_frag((const __bf16*)Ks + (j * 16) * LDT + kk, LDT);
                s4[j] = wmma_bf16(aq, bk, s4[j]);
            }
        }

        const bool diag = (kt == qt);
        float corr[8];
#pragma unroll
        for (int r = 0; r < 8; ++r) {
            const int qglob = q0 + w * 16 + r + 8 * hl;
            float mx = -1e30f;
#pragma unroll
            for (int j = 0; j < 4; ++j) {
                float v = s4[j][r] * scale;
                if (diag && (k0 + j * 16 + nl) > qglob) v = -1e30f;   // causal mask
                s4[j][r] = v;
                mx = fmaxf(mx, v);
            }
            mx = fmaxf(mx, __shfl_xor(mx, 1, 32));
            mx = fmaxf(mx, __shfl_xor(mx, 2, 32));
            mx = fmaxf(mx, __shfl_xor(mx, 4, 32));
            mx = fmaxf(mx, __shfl_xor(mx, 8, 32));
            const float mnew = fmaxf(row_m[r], mx);
            corr[r]  = __expf(row_m[r] - mnew);
            row_m[r] = mnew;

            float sum = 0.0f;
#pragma unroll
            for (int j = 0; j < 4; ++j) {
                const float p = __expf(s4[j][r] - mnew);
                sum += p;
                Ps[(w * 16 + r + 8 * hl) * LDT + j * 16 + nl] = f32_to_bf16u(p);
            }
            sum += __shfl_xor(sum, 1, 32);
            sum += __shfl_xor(sum, 2, 32);
            sum += __shfl_xor(sum, 4, 32);
            sum += __shfl_xor(sum, 8, 32);
            row_l[r] = row_l[r] * corr[r] + sum;
#pragma unroll
            for (int j = 0; j < 4; ++j) o[j][r] *= corr[r];
        }

        // O += P(16x64) @ V(64x64).  Ps slab is wave-private; DS ops are
        // in-order per wave, compiler inserts s_wait_dscnt before reuse.
#pragma unroll
        for (int kk = 0; kk < 64; kk += 32) {
            v16bf ap = load_a_frag((const __bf16*)Ps + (w * 16) * LDT + kk, LDT);
#pragma unroll
            for (int j = 0; j < 4; ++j) {
                v16bf bv = load_b_frag((const __bf16*)Vts + (j * 16) * LDT + kk, LDT);
                o[j] = wmma_bf16(ap, bv, o[j]);
            }
        }
    }

    // normalize and write Y (B,T,C) as bf16
#pragma unroll
    for (int j = 0; j < 4; ++j)
#pragma unroll
        for (int r = 0; r < 8; ++r) {
            const int m = w * 16 + r + 8 * hl;
            const float v = o[j][r] / row_l[r];
            Yg[((size_t)(b * 2048 + q0 + m)) * 1024 + h * 64 + j * 16 + nl] = f32_to_bf16u(v);
        }
}

// ------------------------------ launcher -----------------------------------

extern "C" void kernel_launch(void* const* d_in, const int* in_sizes, int n_in,
                              void* d_out, int out_size, void* d_ws, size_t ws_size,
                              hipStream_t stream) {
    (void)in_sizes; (void)n_in; (void)out_size; (void)ws_size;
    const int Bsz = 4, T = 2048, C = 1024, H = 16;
    const int M = Bsz * T;                       // 8192

    const float* x  = (const float*)d_in[0];
    const float* Wq = (const float*)d_in[1]; const float* bq = (const float*)d_in[2];
    const float* Wk = (const float*)d_in[3]; const float* bk = (const float*)d_in[4];
    const float* Wv = (const float*)d_in[5]; const float* bv = (const float*)d_in[6];
    const float* Wp = (const float*)d_in[7]; const float* bp = (const float*)d_in[8];

    unsigned short* ws = (unsigned short*)d_ws;
    size_t off = 0;
    unsigned short* xb  = ws + off; off += (size_t)M * C;
    unsigned short* wqt = ws + off; off += (size_t)C * C;
    unsigned short* wkt = ws + off; off += (size_t)C * C;
    unsigned short* wvt = ws + off; off += (size_t)C * C;
    unsigned short* wpt = ws + off; off += (size_t)C * C;
    unsigned short* Qb  = ws + off; off += (size_t)M * C;
    unsigned short* Kb  = ws + off; off += (size_t)M * C;
    unsigned short* Vb  = ws + off; off += (size_t)M * C;
    unsigned short* Yb  = ws + off; off += (size_t)M * C;   // ~92 MB total

    k_cvt_bf16<<<(M * C + 255) / 256, 256, 0, stream>>>(x, xb, M * C);
    k_cvt_transpose<<<(C * C + 255) / 256, 256, 0, stream>>>(Wq, wqt, C, C);
    k_cvt_transpose<<<(C * C + 255) / 256, 256, 0, stream>>>(Wk, wkt, C, C);
    k_cvt_transpose<<<(C * C + 255) / 256, 256, 0, stream>>>(Wv, wvt, C, C);
    k_cvt_transpose<<<(C * C + 255) / 256, 256, 0, stream>>>(Wp, wpt, C, C);

    dim3 gg(C / 64, M / 128);
    k_gemm<unsigned short><<<gg, 256, 0, stream>>>(xb, wqt, bq, Qb, M, C, C);
    k_gemm<unsigned short><<<gg, 256, 0, stream>>>(xb, wkt, bk, Kb, M, C, C);
    k_gemm<unsigned short><<<gg, 256, 0, stream>>>(xb, wvt, bv, Vb, M, C, C);

    k_attn<<<dim3(T / 64, Bsz * H), 128, 0, stream>>>(Qb, Kb, Vb, Yb);

    k_gemm<float><<<gg, 256, 0, stream>>>(Yb, wpt, bp, (float*)d_out, M, C, C);
}
